// DistributedMultiheadDilatedAttention_65755949301794
// MI455X (gfx1250) — compile-verified, matrix operand order checked by reading, not run on hardware
//
#include <hip/hip_runtime.h>

typedef __attribute__((ext_vector_type(16))) _Float16 v16h;
typedef __attribute__((ext_vector_type(8)))  float    v8f;

#define EMBED    1024
#define HEADS    16
#define HEAD_DIM 64
#define NTOK     8192

#define WMMA_F16(a, b, c) \
  __builtin_amdgcn_wmma_f32_16x16x32_f16(false, (a), false, (b), (short)0, (c), false, false)

// ---- gfx1250 async global->LDS copy (ASYNCcnt path, ISA ch.10.7/15.18.3) ---
__device__ __forceinline__ void async_to_lds_32B(void* lds, const void* gsrc) {
  const unsigned ldst = (unsigned)(unsigned long long)lds;
  asm volatile(
      "global_load_async_to_lds_b128 %0, %1, off\n\t"
      "global_load_async_to_lds_b128 %0, %1, off offset:16"
      :
      : "v"(ldst), "v"(gsrc)
      : "memory");
}

__device__ __forceinline__ void wait_asynccnt0() {
#if __has_builtin(__builtin_amdgcn_s_wait_asynccnt)
  __builtin_amdgcn_s_wait_asynccnt(0);
#else
  asm volatile("s_wait_asynccnt 0x0" ::: "memory");
#endif
}

__device__ __forceinline__ void wait_tensorcnt0() {
#if __has_builtin(__builtin_amdgcn_s_wait_tensorcnt)
  __builtin_amdgcn_s_wait_tensorcnt(0);
#else
  asm volatile("s_wait_tensorcnt 0x0" ::: "memory");
#endif
}

// ---- gfx1250 Tensor Data Mover: 2D tile DMA into LDS (TENSORcnt path) -----
// Builds a D# (ISA 08 §8.3/8.4): 2-byte elements, tile 64x32, row stride in
// elements, LDS padding 4 DWORDs every 32 DWORDs -> 144 B LDS row pitch (=72
// halfs), matching the WMMA staging layout.
#if __has_builtin(__builtin_amdgcn_tensor_load_to_lds)
#define HAVE_TDM 1
typedef __attribute__((ext_vector_type(4))) unsigned tdm_v4u;
typedef __attribute__((ext_vector_type(8))) int      tdm_v8i;
typedef __attribute__((ext_vector_type(4))) int      tdm_v4i;

__device__ __forceinline__ void tdm_load_tile_64x32(
    void* lds_dst, const void* gsrc, int row_stride_elems) {
  const unsigned long long ga = (unsigned long long)gsrc;
  tdm_v4u g0;
  g0[0] = 1u;                                              // count=1 (valid D#)
  g0[1] = (unsigned)(unsigned long long)lds_dst;           // lds_addr (bytes)
  g0[2] = (unsigned)(ga & 0xFFFFFFFFull);                  // global_addr[31:0]
  g0[3] = (unsigned)((ga >> 32) & 0x1FFFFFFull) | (2u << 30);  // [56:32]|type=2

  tdm_v8i g1;
  g1[0] = (1 << 16)        // data_size = 2 bytes
        | (1 << 20)        // pad_enable
        | (4 << 22)        // pad_interval: 32 DWORDs (128 B row) before pad
        | (3 << 25);       // pad_amount: 4 DWORDs (16 B) -> 144 B pitch
  g1[1] = (64 << 16);      // [47:32] barrier addr = 0 | tensor_dim0[15:0] = 64
  g1[2] = (32 << 16);      // tensor_dim0[31:16]=0 | tensor_dim1[15:0] = 32
  g1[3] = (64 << 16);      // tensor_dim1[31:16]=0 | tile_dim0 = 64
  g1[4] = 32;              // tile_dim1 = 32 | tile_dim2 = 0
  g1[5] = row_stride_elems;  // tensor_dim0_stride[31:0] (data_size units)
  g1[6] = 0;               // stride[47:32] | tensor_dim1_stride[15:0]
  g1[7] = 0;

  const tdm_v4i z4 = {};
#if defined(__clang_major__) && __clang_major__ >= 23
  const tdm_v8i z8 = {};
  __builtin_amdgcn_tensor_load_to_lds(g0, g1, z4, z4, z8, 0);
#else
  __builtin_amdgcn_tensor_load_to_lds(g0, g1, z4, z4, 0);
#endif
}
#else
#define HAVE_TDM 0
#endif

// ---- 16-lane butterfly reductions (v_permlane16_b32) -----------------------
#if __has_builtin(__builtin_amdgcn_permlane16)
#define HAVE_PERMLANE16 1
__device__ __forceinline__ float permx16(float x, unsigned lo, unsigned hi) {
  const int xi = __float_as_int(x);
  return __int_as_float(
      __builtin_amdgcn_permlane16(xi, xi, (int)lo, (int)hi, false, false));
}
#else
#define HAVE_PERMLANE16 0
#endif

__device__ __forceinline__ float rowmax16(float x) {
#if HAVE_PERMLANE16
  x = fmaxf(x, permx16(x, 0x67452301u, 0xEFCDAB89u));  // xor 1
  x = fmaxf(x, permx16(x, 0x54761032u, 0xDCFE98BAu));  // xor 2
  x = fmaxf(x, permx16(x, 0x32107654u, 0xBA98FEDCu));  // xor 4
  x = fmaxf(x, permx16(x, 0xFEDCBA98u, 0x76543210u));  // xor 8
#else
  x = fmaxf(x, __shfl_xor(x, 1, 32));
  x = fmaxf(x, __shfl_xor(x, 2, 32));
  x = fmaxf(x, __shfl_xor(x, 4, 32));
  x = fmaxf(x, __shfl_xor(x, 8, 32));
#endif
  return x;
}

__device__ __forceinline__ float rowsum16(float x) {
#if HAVE_PERMLANE16
  x += permx16(x, 0x67452301u, 0xEFCDAB89u);
  x += permx16(x, 0x54761032u, 0xDCFE98BAu);
  x += permx16(x, 0x32107654u, 0xBA98FEDCu);
  x += permx16(x, 0xFEDCBA98u, 0x76543210u);
#else
  x += __shfl_xor(x, 1, 32);
  x += __shfl_xor(x, 2, 32);
  x += __shfl_xor(x, 4, 32);
  x += __shfl_xor(x, 8, 32);
#endif
  return x;
}

// ---- WMMA fragment loaders (CDNA5 ISA 7.12.2 layouts) ----------------------
__device__ __forceinline__ v16h wmma_load_a(const _Float16* p, int ld) {
  const int lane = threadIdx.x & 31;
  const int row = lane & 15, hi = lane >> 4;
  v16h a;
#pragma unroll
  for (int r = 0; r < 8; ++r) {
    const int k = ((r & 3) << 1) + ((r >> 2) << 4) + (hi << 3);
    a[2 * r]     = p[row * ld + k];
    a[2 * r + 1] = p[row * ld + k + 1];
  }
  return a;
}

__device__ __forceinline__ v16h wmma_load_b_nk(const _Float16* p, int ld) {
  const int lane = threadIdx.x & 31;
  const int n = lane & 15, hi = lane >> 4;
  v16h b;
#pragma unroll
  for (int r = 0; r < 8; ++r) {
    const int k = ((r & 3) << 1) + ((r >> 2) << 4) + (hi << 3);
    b[2 * r]     = p[n * ld + k];
    b[2 * r + 1] = p[n * ld + k + 1];
  }
  return b;
}

__device__ __forceinline__ v16h wmma_load_b_kn(const _Float16* p, int ld) {
  const int lane = threadIdx.x & 31;
  const int n = lane & 15, hi = lane >> 4;
  v16h b;
#pragma unroll
  for (int r = 0; r < 8; ++r) {
    const int k = ((r & 3) << 1) + ((r >> 2) << 4) + (hi << 3);
    b[2 * r]     = p[k * ld + n];
    b[2 * r + 1] = p[(k + 1) * ld + n];
  }
  return b;
}

// ---- GEMM: C[i][j] = sum_k X[i][k] * W[j][k] + bias[j] ---------------------
template <bool OUT_HALF>
__global__ __launch_bounds__(128) void gemm_bias_kernel(
    const float* __restrict__ X, const float* __restrict__ W,
    const float* __restrict__ bias, void* __restrict__ out) {
  __shared__ _Float16 At[64][40];
  __shared__ _Float16 Bt[64][40];
  const int tid = threadIdx.x;
  const int wave = tid >> 5, lane = tid & 31;
  const int hi = lane >> 4, nn = lane & 15;
  const int rowbase = blockIdx.y * 64;
  const int colbase = blockIdx.x * 64;

  v8f acc[4] = {};
  for (int kk = 0; kk < EMBED; kk += 32) {
#pragma unroll
    for (int e = 0; e < 16; ++e) {
      const int flat = tid * 16 + e;  // 2048 elems / 128 threads
      const int r = flat >> 5, c = flat & 31;
      At[r][c] = (_Float16)X[(size_t)(rowbase + r) * EMBED + kk + c];
      Bt[r][c] = (_Float16)W[(size_t)(colbase + r) * EMBED + kk + c];
    }
    // pull next K-chunk toward the WGP while this chunk computes
    if (kk + 32 < EMBED) {
      const int pr = tid >> 1, pc = (tid & 1) * 16;
      __builtin_prefetch(&X[(size_t)(rowbase + pr) * EMBED + kk + 32 + pc], 0, 0);
      __builtin_prefetch(&W[(size_t)(colbase + pr) * EMBED + kk + 32 + pc], 0, 0);
    }
    __syncthreads();
    const v16h a = wmma_load_a(&At[wave * 16][0], 40);
#pragma unroll
    for (int c = 0; c < 4; ++c) {
      const v16h b = wmma_load_b_nk(&Bt[c * 16][0], 40);
      acc[c] = WMMA_F16(a, b, acc[c]);
    }
    __syncthreads();
  }
#pragma unroll
  for (int c = 0; c < 4; ++c)
#pragma unroll
    for (int v = 0; v < 8; ++v) {
      const int row = rowbase + wave * 16 + v + 8 * hi;
      const int col = colbase + c * 16 + nn;
      const float val = acc[c][v] + bias[col];
      if (OUT_HALF)
        ((_Float16*)out)[(size_t)row * EMBED + col] = (_Float16)val;
      else
        ((float*)out)[(size_t)row * EMBED + col] = val;
    }
}

// ---- Flash attention over one dilated group --------------------------------
// Each (segment, head) is a 2048x2048x64 block. WG = 128 threads = 4 waves;
// each wave owns 16 query rows. Double-buffered tiles: K via TDM
// tensor_load_to_lds (TENSORcnt), V via global_load_async_to_lds_b128
// (ASYNCcnt) -- both CDNA5 DMA paths overlap the WMMA compute.
__global__ __launch_bounds__(128) void attn_kernel(
    const _Float16* __restrict__ Q, const _Float16* __restrict__ K,
    const _Float16* __restrict__ V, float* __restrict__ out,
    int seglen, int rate, int off, int hmin, int gheads) {
  __shared__ _Float16 Kt[2][32][72];
  __shared__ _Float16 Vt[2][32][72];
  __shared__ _Float16 Pt[4][16][34];

  const int S = seglen / rate;  // 2048 dilated positions per segment
  const int qtiles = S / 64;
  const int bid = blockIdx.x;
  const int qt = bid % qtiles;
  const int h = (bid / qtiles) % gheads;
  const int seg = bid / (qtiles * gheads);
  const int head = hmin + h;
  const int segbase = seg * seglen + off;

  const int tid = threadIdx.x, wave = tid >> 5, lane = tid & 31;
  const int hi = lane >> 4, nn = lane & 15;
  const size_t hoff = (size_t)head * HEAD_DIM;

  // per-thread async staging slot: quarter-row of the 32x64 tile (32 B)
  const int srow = tid >> 2, scol = (tid & 3) * 16;

  // Q fragments for this wave's 16 query rows (A-layout gather from global)
  const int qrowA = qt * 64 + wave * 16 + nn;
  const _Float16* qp = Q + (size_t)(segbase + qrowA * rate) * EMBED + hoff;
  v16h qa0, qa1;
#pragma unroll
  for (int r = 0; r < 8; ++r) {
    const int k = ((r & 3) << 1) + ((r >> 2) << 4) + (hi << 3);
    qa0[2 * r]     = qp[k];
    qa0[2 * r + 1] = qp[k + 1];
    qa1[2 * r]     = qp[32 + k];
    qa1[2 * r + 1] = qp[32 + k + 1];
  }

  v8f o0 = {}, o1 = {}, o2 = {}, o3 = {};
  float row_m[8], row_l[8];
#pragma unroll
  for (int v = 0; v < 8; ++v) { row_m[v] = -1e30f; row_l[v] = 0.f; }
  const float scale = 0.125f;  // 1/sqrt(64)

  // issue DMA for tile 0
  {
#if HAVE_TDM
    if (wave == 0)
      tdm_load_tile_64x32(&Kt[0][0][0], K + (size_t)segbase * EMBED + hoff,
                          rate * EMBED);
#else
    {
      const size_t kpos = (size_t)(segbase + srow * rate);
      async_to_lds_32B(&Kt[0][srow][scol], K + kpos * EMBED + hoff + scol);
    }
#endif
    const size_t vpos = (size_t)(segbase + srow * rate);
    async_to_lds_32B(&Vt[0][srow][scol], V + vpos * EMBED + hoff + scol);
  }

  for (int kb = 0; kb < S; kb += 32) {
    const int buf = (kb >> 5) & 1;
    wait_asynccnt0();  // this wave's V-tile DMA has landed in LDS
#if HAVE_TDM
    if (wave == 0) wait_tensorcnt0();  // K-tile TDM has landed
#endif
    __syncthreads();   // ... visible to every wave

    // overlap: issue DMA for the next tile into the other buffer
    if (kb + 32 < S) {
#if HAVE_TDM
      if (wave == 0)
        tdm_load_tile_64x32(&Kt[buf ^ 1][0][0],
                            K + (size_t)(segbase + (kb + 32) * rate) * EMBED + hoff,
                            rate * EMBED);
#else
      {
        const size_t kpos = (size_t)(segbase + (kb + 32 + srow) * rate);
        async_to_lds_32B(&Kt[buf ^ 1][srow][scol],
                         K + kpos * EMBED + hoff + scol);
      }
#endif
      const size_t vpos = (size_t)(segbase + (kb + 32 + srow) * rate);
      async_to_lds_32B(&Vt[buf ^ 1][srow][scol], V + vpos * EMBED + hoff + scol);
    }

    // S = Q K^T : two 16x16 C-tiles, d=64 -> 2 accumulation steps each
    v8f s0 = {}, s1 = {};
    {
      v16h b;
      b = wmma_load_b_nk(&Kt[buf][0][0], 72);   s0 = WMMA_F16(qa0, b, s0);
      b = wmma_load_b_nk(&Kt[buf][0][32], 72);  s0 = WMMA_F16(qa1, b, s0);
      b = wmma_load_b_nk(&Kt[buf][16][0], 72);  s1 = WMMA_F16(qa0, b, s1);
      b = wmma_load_b_nk(&Kt[buf][16][32], 72); s1 = WMMA_F16(qa1, b, s1);
    }

    // online softmax (f32); row = v + 8*hi, 16 cols spread over 16 lanes
    float p0[8], p1[8], alpha[8];
#pragma unroll
    for (int v = 0; v < 8; ++v) {
      const float a = s0[v] * scale, c = s1[v] * scale;
      const float mx = rowmax16(fmaxf(a, c));
      const float mnew = fmaxf(row_m[v], mx);
      alpha[v] = __expf(row_m[v] - mnew);
      row_m[v] = mnew;
      p0[v] = __expf(a - mnew);
      p1[v] = __expf(c - mnew);
      row_l[v] = row_l[v] * alpha[v] + rowsum16(p0[v] + p1[v]);
    }

    // rescale O accumulators; stage P as f16 (C-layout -> row-major LDS)
#pragma unroll
    for (int v = 0; v < 8; ++v) {
      o0[v] *= alpha[v]; o1[v] *= alpha[v];
      o2[v] *= alpha[v]; o3[v] *= alpha[v];
      const int prow = v + 8 * hi;
      Pt[wave][prow][nn]      = (_Float16)p0[v];
      Pt[wave][prow][nn + 16] = (_Float16)p1[v];
    }
    __syncthreads();

    // O += P V : A = P (16x32), B = V 32x16 chunks of d
    {
      const v16h pa = wmma_load_a(&Pt[wave][0][0], 34);
      v16h b;
      b = wmma_load_b_kn(&Vt[buf][0][0], 72);  o0 = WMMA_F16(pa, b, o0);
      b = wmma_load_b_kn(&Vt[buf][0][16], 72); o1 = WMMA_F16(pa, b, o1);
      b = wmma_load_b_kn(&Vt[buf][0][32], 72); o2 = WMMA_F16(pa, b, o2);
      b = wmma_load_b_kn(&Vt[buf][0][48], 72); o3 = WMMA_F16(pa, b, o3);
    }
  }

  // epilogue: scatter O/l to dilated global positions
#pragma unroll
  for (int v = 0; v < 8; ++v) {
    const int qr = qt * 64 + wave * 16 + v + 8 * hi;
    const size_t pos = (size_t)(segbase + qr * rate);
    const float inv = 1.0f / row_l[v];
    float* op = out + pos * EMBED + hoff;
    op[nn]      = o0[v] * inv;
    op[16 + nn] = o1[v] * inv;
    op[32 + nn] = o2[v] * inv;
    op[48 + nn] = o3[v] * inv;
  }
}

// ---- helpers ---------------------------------------------------------------
__global__ void zero_f32(float* __restrict__ p, long n) {
  long i = (long)blockIdx.x * blockDim.x + threadIdx.x;
  const long stride = (long)gridDim.x * blockDim.x;
  for (; i < n; i += stride) p[i] = 0.f;
}

__global__ __launch_bounds__(256) void layernorm_inplace(
    float* __restrict__ x, const float* __restrict__ g,
    const float* __restrict__ b) {
  __shared__ float s_sum[256];
  __shared__ float s_sq[256];
  const int row = blockIdx.x, tid = threadIdx.x;
  float* xr = x + (size_t)row * EMBED;
  float v[4], s = 0.f, sq = 0.f;
#pragma unroll
  for (int i = 0; i < 4; ++i) {
    v[i] = xr[tid + i * 256];
    s += v[i];
    sq += v[i] * v[i];
  }
  s_sum[tid] = s;
  s_sq[tid] = sq;
  __syncthreads();
  for (int st = 128; st > 0; st >>= 1) {
    if (tid < st) {
      s_sum[tid] += s_sum[tid + st];
      s_sq[tid] += s_sq[tid + st];
    }
    __syncthreads();
  }
  const float mean = s_sum[0] * (1.f / EMBED);
  const float var = s_sq[0] * (1.f / EMBED) - mean * mean;
  const float rstd = rsqrtf(var + 1e-5f);
#pragma unroll
  for (int i = 0; i < 4; ++i) {
    const int c = tid + i * 256;
    xr[c] = (v[i] - mean) * rstd * g[c] + b[c];
  }
}

// ---- launcher --------------------------------------------------------------
extern "C" void kernel_launch(void* const* d_in, const int* in_sizes, int n_in,
                              void* d_out, int out_size, void* d_ws,
                              size_t ws_size, hipStream_t stream) {
  const float* query = (const float*)d_in[0];
  const float* key   = (const float*)d_in[1];
  const float* value = (const float*)d_in[2];
  const float* Wq = (const float*)d_in[3];  const float* bq = (const float*)d_in[4];
  const float* Wk = (const float*)d_in[5];  const float* bk = (const float*)d_in[6];
  const float* Wv = (const float*)d_in[7];  const float* bv = (const float*)d_in[8];
  const float* Wo = (const float*)d_in[9];  const float* bo = (const float*)d_in[10];
  const float* ln_g = (const float*)d_in[11];
  const float* ln_b = (const float*)d_in[12];

  char* ws = (char*)d_ws;
  const size_t nelem = (size_t)NTOK * EMBED;  // 8M elements
  _Float16* qf = (_Float16*)(ws);             // 16 MB
  _Float16* kf = (_Float16*)(ws + nelem * 2); // 16 MB
  _Float16* vf = (_Float16*)(ws + nelem * 4); // 16 MB
  float*  attn = (float*)(ws + nelem * 6);    // 32 MB

  const dim3 gblk(EMBED / 64, NTOK / 64);     // (16, 128)
  gemm_bias_kernel<true><<<gblk, 128, 0, stream>>>(query, Wq, bq, qf);
  gemm_bias_kernel<true><<<gblk, 128, 0, stream>>>(key,   Wk, bk, kf);
  gemm_bias_kernel<true><<<gblk, 128, 0, stream>>>(value, Wv, bv, vf);

  zero_f32<<<2048, 256, 0, stream>>>(attn, (long)nelem);

  // group 0: seg 2048, r=1, off=0, heads [0,6)
  attn_kernel<<<4 * 6 * 32, 128, 0, stream>>>(qf, kf, vf, attn, 2048, 1, 0, 0, 6);
  // group 1: seg 4096, r=2, off=1, heads [6,11)
  attn_kernel<<<2 * 5 * 32, 128, 0, stream>>>(qf, kf, vf, attn, 4096, 2, 1, 6, 5);
  // group 2: seg 8192, r=4, off=2, heads [11,16)
  attn_kernel<<<1 * 5 * 32, 128, 0, stream>>>(qf, kf, vf, attn, 8192, 4, 2, 11, 5);

  layernorm_inplace<<<NTOK, 256, 0, stream>>>(attn, ln_g, ln_b);

  gemm_bias_kernel<false><<<gblk, 128, 0, stream>>>(attn, Wo, bo, (float*)d_out);
}
